// SRUCell_3882650436170
// MI455X (gfx1250) — matrix operand order repaired and checked by reading, not compiled, not run on hardware
//
#include <hip/hip_runtime.h>

typedef __attribute__((ext_vector_type(16))) __bf16 v16bf;
typedef __attribute__((ext_vector_type(8)))  float  v8f;
typedef __attribute__((ext_vector_type(4)))  float  v4f;

#define L_TOTAL 2048
#define BATCH   32
#define DDIM    1024
#define NDIM    3072                 // 3*D
#define KDIM    1024                 // D
#define CHUNK_L 256
#define CHUNK_M (CHUNK_L * BATCH)    // 8192 rows per GEMM chunk
#define NCHUNKS (L_TOTAL / CHUNK_L)  // 8

// ---------------------------------------------------------------------------
// Transpose + convert weight (K=1024, N=3072) f32 row-major ->
// Wt (N, K) bf16 row-major, so B fragments are contiguous per lane.
// ---------------------------------------------------------------------------
__global__ __launch_bounds__(256)
void sru_wt_kernel(const float* __restrict__ w, __bf16* __restrict__ wt) {
  __shared__ float tile[32][33];
  const int k0 = blockIdx.x * 32;
  const int n0 = blockIdx.y * 32;
  const int tx = threadIdx.x & 31;
  const int ty = threadIdx.x >> 5;            // 0..7
#pragma unroll
  for (int i = 0; i < 32; i += 8)
    tile[ty + i][tx] = w[(size_t)(k0 + ty + i) * NDIM + (n0 + tx)];
  __syncthreads();
#pragma unroll
  for (int i = 0; i < 32; i += 8)
    wt[(size_t)(n0 + ty + i) * KDIM + (k0 + tx)] = (__bf16)tile[tx][ty + i];
}

// ---------------------------------------------------------------------------
// GEMM chunk: u[CHUNK_M x NDIM] (f32) = x_chunk[CHUNK_M x KDIM] (f32->bf16)
//             @ Wt^T, via v_wmma_f32_16x16x32_bf16.
// Block = 128 threads (4 waves). Block tile = 64(M) x 64(N).
// Each wave: 16(M) x 64(N) strip = 1 A fragment, 4 B fragments, 4 accums.
// ---------------------------------------------------------------------------
__global__ __launch_bounds__(128)
void sru_gemm_kernel(const float* __restrict__ x, const __bf16* __restrict__ wt,
                     float* __restrict__ u) {
  const int lane   = threadIdx.x & 31;
  const int wave   = threadIdx.x >> 5;   // 0..3
  const int laneLo = lane & 15;
  const int hi     = lane >> 4;          // 0 or 1

  const int m0 = blockIdx.x * 64 + wave * 16;
  const int n0 = blockIdx.y * 64;

  // A k order used here (contiguous, same as B): lanes 0-15 hold row M=lane, K = 0..15;
  // lanes 16-31 hold row M=lane-16, K = 16..31.
  const float* arow = x + (size_t)(m0 + laneLo) * KDIM + hi * 16;

  // B k order used here (contiguous, same as A): lanes 0-15 hold col N=lane, K=0..15 contiguous;
  // lanes 16-31 hold K=16..31.  Wt is (N,K) row-major -> contiguous 32B/lane.
  const __bf16* b0p = wt + (size_t)(n0 +  0 + laneLo) * KDIM + hi * 16;
  const __bf16* b1p = wt + (size_t)(n0 + 16 + laneLo) * KDIM + hi * 16;
  const __bf16* b2p = wt + (size_t)(n0 + 32 + laneLo) * KDIM + hi * 16;
  const __bf16* b3p = wt + (size_t)(n0 + 48 + laneLo) * KDIM + hi * 16;

  v8f acc0 = {}, acc1 = {}, acc2 = {}, acc3 = {};

  for (int k0 = 0; k0 < KDIM; k0 += 32) {
    v4f a0 = *(const v4f*)(arow + k0);        // K  +0..3
    v4f a1 = *(const v4f*)(arow + k0 + 4);    // K  +4..7
    v4f a2 = *(const v4f*)(arow + k0 + 8);    // K  +8..11
    v4f a3 = *(const v4f*)(arow + k0 + 12);   // K +12..15
    v16bf a;
#pragma unroll
    for (int i = 0; i < 4; ++i) {
      a[i]      = (__bf16)a0[i];
      a[4 + i]  = (__bf16)a1[i];
      a[8 + i]  = (__bf16)a2[i];
      a[12 + i] = (__bf16)a3[i];
    }
    v16bf b0 = *(const v16bf*)(b0p + k0);
    v16bf b1 = *(const v16bf*)(b1p + k0);
    v16bf b2 = *(const v16bf*)(b2p + k0);
    v16bf b3 = *(const v16bf*)(b3p + k0);

    acc0 = __builtin_amdgcn_wmma_f32_16x16x32_bf16(false, a, false, b0, (short)0, acc0, false, false);
    acc1 = __builtin_amdgcn_wmma_f32_16x16x32_bf16(false, a, false, b1, (short)0, acc1, false, false);
    acc2 = __builtin_amdgcn_wmma_f32_16x16x32_bf16(false, a, false, b2, (short)0, acc2, false, false);
    acc3 = __builtin_amdgcn_wmma_f32_16x16x32_bf16(false, a, false, b3, (short)0, acc3, false, false);
  }

  // C/D 16x16 f32 layout: VGPR r, lanes 0-15 -> (M=r, N=lane);
  // lanes 16-31 -> (M=8+r, N=lane-16).
  float* urow = u + (size_t)(m0 + hi * 8) * NDIM + n0 + laneLo;
#pragma unroll
  for (int r = 0; r < 8; ++r) {
    urow[(size_t)r * NDIM +  0] = acc0[r];
    urow[(size_t)r * NDIM + 16] = acc1[r];
    urow[(size_t)r * NDIM + 32] = acc2[r];
    urow[(size_t)r * NDIM + 48] = acc3[r];
  }
}

// ---------------------------------------------------------------------------
// Sequential scan over one chunk of CHUNK_L timesteps.
// One thread per (b, d) channel; u chunk expected hot in L2.
// ---------------------------------------------------------------------------
__device__ __forceinline__ float sigmoidf_dev(float z) {
  return 1.0f / (1.0f + __expf(-z));
}

__global__ __launch_bounds__(256)
void sru_scan_kernel(const float* __restrict__ x, const float* __restrict__ u,
                     const float* __restrict__ wc, const float* __restrict__ bias,
                     const float* __restrict__ scale_x,
                     float* __restrict__ h, float* __restrict__ carry,
                     float* __restrict__ cfinal, int l0, int first) {
  const int idx = blockIdx.x * 256 + threadIdx.x;  // idx = b*DDIM + d
  const int d = idx & (DDIM - 1);
  const int b = idx >> 10;

  const float fw = wc[d];
  const float rw = wc[DDIM + d];
  const float fb = bias[d];
  const float rb = bias[DDIM + d];
  const float sx = scale_x[0];

  float c = first ? 0.0f : carry[idx];

  size_t xoff = ((size_t)l0 * BATCH + b) * DDIM + d;       // into full x / h
  size_t uoff = (size_t)b * NDIM + 3 * (size_t)d;          // into u chunk

  for (int t = 0; t < CHUNK_L; ++t) {
    const float u0 = u[uoff];
    const float u1 = u[uoff + 1] + fb;
    const float u2 = u[uoff + 2] + rb;
    const float xv = x[xoff] * sx;

    const float f = sigmoidf_dev(u1 + c * fw);
    const float r = sigmoidf_dev(u2 + c * rw);
    c = u0 + (c - u0) * f;
    h[xoff] = xv + (c - xv) * r;

    xoff += (size_t)BATCH * DDIM;
    uoff += (size_t)BATCH * NDIM;
  }
  carry[idx] = c;
  if (cfinal) cfinal[idx] = c;
}

// ---------------------------------------------------------------------------
extern "C" void kernel_launch(void* const* d_in, const int* in_sizes, int n_in,
                              void* d_out, int out_size, void* d_ws, size_t ws_size,
                              hipStream_t stream) {
  (void)in_sizes; (void)n_in; (void)out_size; (void)ws_size;

  const float* x       = (const float*)d_in[0];
  const float* weight  = (const float*)d_in[1];
  const float* wc      = (const float*)d_in[2];
  const float* bias    = (const float*)d_in[3];
  const float* scale_x = (const float*)d_in[4];

  float* h      = (float*)d_out;
  float* cfinal = h + (size_t)L_TOTAL * BATCH * DDIM;

  // Workspace layout: [Wt bf16 | u chunk f32 | carry f32]
  char*   ws     = (char*)d_ws;
  __bf16* wt     = (__bf16*)ws;
  size_t  wt_b   = (size_t)NDIM * KDIM * sizeof(__bf16);    // 6 MB, 256B-mult
  float*  uchunk = (float*)(ws + wt_b);
  float*  carry  = uchunk + (size_t)CHUNK_M * NDIM;         // 96 MB later

  sru_wt_kernel<<<dim3(KDIM / 32, NDIM / 32), 256, 0, stream>>>(weight, wt);

  for (int chunk = 0; chunk < NCHUNKS; ++chunk) {
    const int l0 = chunk * CHUNK_L;
    const float* xc = x + (size_t)l0 * BATCH * DDIM;

    sru_gemm_kernel<<<dim3(CHUNK_M / 64, NDIM / 64), 128, 0, stream>>>(xc, wt, uchunk);

    sru_scan_kernel<<<dim3((BATCH * DDIM) / 256), 256, 0, stream>>>(
        x, uchunk, wc, bias, scale_x, h, carry,
        (chunk == NCHUNKS - 1) ? cfinal : nullptr, l0, chunk == 0 ? 1 : 0);
  }
}
